// MultiHeadAttention_21784074125593
// MI455X (gfx1250) — matrix operand order repaired and checked by reading, not compiled, not run on hardware
//
#include <hip/hip_runtime.h>
#include <stdint.h>

#define D_MODEL 1024
#define NUM_HEADS 16
#define D_K 64
#define BATCH 2
#define SEQ 2048
#define MROWS (BATCH * SEQ) /* 4096 */

typedef __attribute__((ext_vector_type(16))) __bf16 v16bf;
typedef __attribute__((ext_vector_type(8)))  __bf16 v8bf;
typedef __attribute__((ext_vector_type(4)))  __bf16 v4bf;
typedef __attribute__((ext_vector_type(8)))  float  v8f;

__device__ __forceinline__ __bf16 f2bf(float f) {
  union { float f; uint32_t u; } v; v.f = f;
  uint32_t r = v.u + 0x7FFFu + ((v.u >> 16) & 1u);  // round-to-nearest-even
  union { uint16_t u; __bf16 h; } o; o.u = (uint16_t)(r >> 16);
  return o.h;
}

__device__ __forceinline__ v16bf cat16(v8bf lo, v8bf hi) {
  v16bf r;
#pragma unroll
  for (int i = 0; i < 8; ++i) { r[i] = lo[i]; r[i + 8] = hi[i]; }
  return r;
}

// A fragment (16 rows x 32 K, bf16) from row-major [16][ld] tile.
__device__ __forceinline__ v16bf load_frag_a(const __bf16* base, int ld, int lane) {
  const int m = lane & 15, hlf = lane >> 4;
  const __bf16* p = base + m * ld + hlf * 8;
  return cat16(*(const v8bf*)p, *(const v8bf*)(p + 16));
}

// B fragment (32 K x 16 n, bf16) from tile stored n-major [16 rows n][ld over K].
__device__ __forceinline__ v16bf load_frag_b(const __bf16* base, int ld, int lane) {
  const int n = lane & 15, hlf = lane >> 4;
  const __bf16* p = base + n * ld + hlf * 8;
  return cat16(*(const v8bf*)p, *(const v8bf*)(p + 16));
}

#define WMMA_BF16(a, b, c) \
  __builtin_amdgcn_wmma_f32_16x16x32_bf16(false, (a), false, (b), (short)0, (c), false, false)

// Async global->LDS 16-byte copy (ASYNCcnt-tracked, CDNA5).
// lds_off = low 32 bits of the shared-memory pointer (LDS aperture offset).
__device__ __forceinline__ void async_ld_b128(uint32_t lds_off, const __bf16* gptr) {
  asm volatile("global_load_async_to_lds_b128 %0, %1, off"
               :: "v"(lds_off), "v"(gptr)
               : "memory");
}
__device__ __forceinline__ void wait_async0() {
  asm volatile("s_wait_asynccnt 0" ::: "memory");
}
__device__ __forceinline__ uint32_t lds_off32(const void* p) {
  return (uint32_t)(uintptr_t)p;  // LDS aperture: addr[31:0] is the DS offset
}

// ---------------------------------------------------------------------------
// Projection GEMM: Y = X @ W^T + bias.  X: [4096][1024] (f32 or bf16),
// W: [1024][1024] f32 row-major.  Block 256 thr / 8 waves, tile M=128 N=64.
// MODE 0: bf16 [B][H][S][64] ; MODE 1: bf16 [B][H][64][S] ; MODE 2: f32 out.
// ---------------------------------------------------------------------------
template <int MODE, bool INBF>
__global__ __launch_bounds__(256) void proj_kernel(
    const float* __restrict__ Xf, const __bf16* __restrict__ Xbf,
    const float* __restrict__ W, const float* __restrict__ bias,
    __bf16* __restrict__ Ybf, float* __restrict__ Yf) {
  __shared__ __align__(16) __bf16 lsA[128 * 32];
  __shared__ __align__(16) __bf16 lsB[64 * 32];

  const int t = threadIdx.x;
  const int lane = t & 31;
  const int wave = t >> 5;
  const int mBase = blockIdx.y * 128;
  const int nBase = blockIdx.x * 64;

  v8f acc[4] = {};

  const int rowA = t >> 1;          // 0..127
  const int colA = (t & 1) * 16;    // 0 / 16
  const int rowB = t >> 2;          // 0..63
  const int colB = (t & 3) * 8;     // 0/8/16/24

  for (int k0 = 0; k0 < D_MODEL; k0 += 32) {
    __syncthreads();
    if (INBF) {
      const __bf16* src = Xbf + (size_t)(mBase + rowA) * D_MODEL + k0 + colA;
      if (k0 + 32 < D_MODEL) __builtin_prefetch(src + 32, 0, 1);
      *(v8bf*)&lsA[rowA * 32 + colA]     = *(const v8bf*)src;
      *(v8bf*)&lsA[rowA * 32 + colA + 8] = *(const v8bf*)(src + 8);
    } else {
      const float* src = Xf + (size_t)(mBase + rowA) * D_MODEL + k0 + colA;
      if (k0 + 32 < D_MODEL) __builtin_prefetch(src + 32, 0, 1);
#pragma unroll
      for (int i = 0; i < 4; ++i) {
        float4 v = *(const float4*)(src + i * 4);
        v4bf hh; hh[0] = f2bf(v.x); hh[1] = f2bf(v.y); hh[2] = f2bf(v.z); hh[3] = f2bf(v.w);
        *(v4bf*)&lsA[rowA * 32 + colA + i * 4] = hh;
      }
    }
    {
      const float* src = W + (size_t)(nBase + rowB) * D_MODEL + k0 + colB;
      if (k0 + 32 < D_MODEL) __builtin_prefetch(src + 32, 0, 1);
#pragma unroll
      for (int i = 0; i < 2; ++i) {
        float4 v = *(const float4*)(src + i * 4);
        v4bf hh; hh[0] = f2bf(v.x); hh[1] = f2bf(v.y); hh[2] = f2bf(v.z); hh[3] = f2bf(v.w);
        *(v4bf*)&lsB[rowB * 32 + colB + i * 4] = hh;
      }
    }
    __syncthreads();

    v16bf aF = load_frag_a(&lsA[(wave * 16) * 32], 32, lane);
#pragma unroll
    for (int nt = 0; nt < 4; ++nt) {
      v16bf bF = load_frag_b(&lsB[(nt * 16) * 32], 32, lane);
      acc[nt] = WMMA_BF16(aF, bF, acc[nt]);
    }
  }

  const int hlf = lane >> 4, nn = lane & 15;
#pragma unroll
  for (int nt = 0; nt < 4; ++nt) {
    const int c = nBase + nt * 16 + nn;
    const float bv = bias[c];
#pragma unroll
    for (int j = 0; j < 8; ++j) {
      const int r = mBase + wave * 16 + hlf * 8 + j;
      const float val = acc[nt][j] + bv;
      if (MODE == 0) {
        const int b = r >> 11, s = r & 2047, h = c >> 6, dk = c & 63;
        Ybf[(((size_t)b * NUM_HEADS + h) * SEQ + s) * D_K + dk] = f2bf(val);
      } else if (MODE == 1) {
        const int b = r >> 11, s = r & 2047, h = c >> 6, dk = c & 63;
        Ybf[(((size_t)b * NUM_HEADS + h) * D_K + dk) * SEQ + s] = f2bf(val);
      } else {
        Yf[(size_t)r * D_MODEL + c] = val;
      }
    }
  }
}

// ---------------------------------------------------------------------------
// Flash attention, 64 keys / iteration, double-buffered async K/V tiles.
// Block = 8 waves; wave owns a 16-query tile of one (b,h).
// ---------------------------------------------------------------------------
__global__ __launch_bounds__(256) void attn_kernel(
    const __bf16* __restrict__ Q, const __bf16* __restrict__ K,
    const __bf16* __restrict__ Vt, const int* __restrict__ mask,
    __bf16* __restrict__ Ctx) {
  __shared__ __align__(16) __bf16 sK[2][64 * 64];   // [buf][key][d]
  __shared__ __align__(16) __bf16 sV[2][64 * 64];   // [buf][dk][key]
  __shared__ __align__(16) __bf16 sP[8 * 16 * 64];  // per-wave P tiles

  const int t = threadIdx.x, lane = t & 31, wave = t >> 5;
  const int bh = blockIdx.y, b = bh >> 4, h = bh & 15;
  const int qBase = (blockIdx.x * 8 + wave) * 16;
  const int hlf = lane >> 4, nn = lane & 15;

  // Q fragments held in registers for the whole key loop (d 0..31, 32..63).
  const __bf16* qp = Q + ((size_t)bh * SEQ + qBase) * D_K;
  const v16bf aQ0 = load_frag_a(qp, D_K, lane);
  const v16bf aQ1 = load_frag_a(qp + 32, D_K, lane);

  v8f acc[4] = {};
  float mi[8], li[8];
#pragma unroll
  for (int j = 0; j < 8; ++j) { mi[j] = -__builtin_huge_valf(); li[j] = 0.f; }

  __bf16* pw = &sP[wave * (16 * 64)];

  // cooperative 64x64 tile mapping: 2 x b128 per thread per tile
  const int rT = t >> 2;            // 0..63
  const int cT = (t & 3) * 16;      // 0/16/32/48

  const float SCALE = 0.125f;               // 1/sqrt(64)
  const float FILL = -3.4028235e38f;        // finfo(f32).min

  // issue async loads of tile 0 into buffer 0
  {
    const __bf16* gK = K + ((size_t)bh * SEQ + 0 + rT) * D_K + cT;
    const __bf16* gV = Vt + ((size_t)bh * D_K + rT) * SEQ + 0 + cT;
    async_ld_b128(lds_off32(&sK[0][rT * 64 + cT]), gK);
    async_ld_b128(lds_off32(&sK[0][rT * 64 + cT + 8]), gK + 8);
    async_ld_b128(lds_off32(&sV[0][rT * 64 + cT]), gV);
    async_ld_b128(lds_off32(&sV[0][rT * 64 + cT + 8]), gV + 8);
  }

  for (int k0 = 0; k0 < SEQ; k0 += 64) {
    const int buf = (k0 >> 6) & 1;
    wait_async0();        // this wave's async loads for `buf` have landed
    __syncthreads();      // everyone's loads landed; prev compute on buf^1 done

    // prefetch next tile into the other buffer, overlapped with compute
    if (k0 + 64 < SEQ) {
      const __bf16* gK = K + ((size_t)bh * SEQ + (k0 + 64) + rT) * D_K + cT;
      const __bf16* gV = Vt + ((size_t)bh * D_K + rT) * SEQ + (k0 + 64) + cT;
      async_ld_b128(lds_off32(&sK[buf ^ 1][rT * 64 + cT]), gK);
      async_ld_b128(lds_off32(&sK[buf ^ 1][rT * 64 + cT + 8]), gK + 8);
      async_ld_b128(lds_off32(&sV[buf ^ 1][rT * 64 + cT]), gV);
      async_ld_b128(lds_off32(&sV[buf ^ 1][rT * 64 + cT + 8]), gV + 8);
    }

    // scores: 16 queries x 64 keys (4 subtiles), K-depth 64 via chained WMMAs
    v8f c[4] = {};
#pragma unroll
    for (int kt = 0; kt < 4; ++kt) {
      c[kt] = WMMA_BF16(aQ0, load_frag_b(&sK[buf][(kt * 16) * 64], 64, lane), c[kt]);
      c[kt] = WMMA_BF16(aQ1, load_frag_b(&sK[buf][(kt * 16) * 64 + 32], 64, lane), c[kt]);
    }

    int mk[4];
#pragma unroll
    for (int kt = 0; kt < 4; ++kt)
      mk[kt] = mask[(size_t)b * SEQ + k0 + kt * 16 + nn];

    // online softmax over 64 keys per iteration
#pragma unroll
    for (int j = 0; j < 8; ++j) {
      float s[4];
#pragma unroll
      for (int kt = 0; kt < 4; ++kt)
        s[kt] = mk[kt] ? c[kt][j] * SCALE : FILL;
      float rm = fmaxf(fmaxf(s[0], s[1]), fmaxf(s[2], s[3]));
#pragma unroll
      for (int off = 1; off < 16; off <<= 1) rm = fmaxf(rm, __shfl_xor(rm, off, 32));
      const float mnew = fmaxf(mi[j], rm);
      const float alpha = __expf(mi[j] - mnew);
      float p[4], rs = 0.f;
#pragma unroll
      for (int kt = 0; kt < 4; ++kt) { p[kt] = __expf(s[kt] - mnew); rs += p[kt]; }
#pragma unroll
      for (int off = 1; off < 16; off <<= 1) rs += __shfl_xor(rs, off, 32);
      li[j] = li[j] * alpha + rs;
      mi[j] = mnew;
#pragma unroll
      for (int kt = 0; kt < 4; ++kt)
        pw[(hlf * 8 + j) * 64 + kt * 16 + nn] = f2bf(p[kt]);
#pragma unroll
      for (int nt = 0; nt < 4; ++nt) acc[nt][j] *= alpha;
    }

    // re-layout P (C-frag -> A-frag) through wave-private LDS
    asm volatile("s_wait_dscnt 0" ::: "memory");
    const v16bf aP0 = load_frag_a(pw, 64, lane);
    const v16bf aP1 = load_frag_a(pw + 32, 64, lane);
#pragma unroll
    for (int nt = 0; nt < 4; ++nt) {
      acc[nt] = WMMA_BF16(aP0, load_frag_b(&sV[buf][(nt * 16) * 64], 64, lane), acc[nt]);
      acc[nt] = WMMA_BF16(aP1, load_frag_b(&sV[buf][(nt * 16) * 64 + 32], 64, lane), acc[nt]);
    }
  }

  // normalize and emit context as bf16 [B][S][D_MODEL]
#pragma unroll
  for (int j = 0; j < 8; ++j) {
    const float inv = 1.0f / li[j];
    const int s = qBase + hlf * 8 + j;
#pragma unroll
    for (int nt = 0; nt < 4; ++nt) {
      Ctx[((size_t)b * SEQ + s) * D_MODEL + h * D_K + nt * 16 + nn] =
          f2bf(acc[nt][j] * inv);
    }
  }
}

extern "C" void kernel_launch(void* const* d_in, const int* in_sizes, int n_in,
                              void* d_out, int out_size, void* d_ws, size_t ws_size,
                              hipStream_t stream) {
  (void)in_sizes; (void)n_in; (void)out_size; (void)ws_size;
  const float* query = (const float*)d_in[0];
  const float* key   = (const float*)d_in[1];
  const float* value = (const float*)d_in[2];
  const int*   mask  = (const int*)d_in[3];
  const float* Wq = (const float*)d_in[4];
  const float* bq = (const float*)d_in[5];
  const float* Wk = (const float*)d_in[6];
  const float* bk = (const float*)d_in[7];
  const float* Wv = (const float*)d_in[8];
  const float* bv = (const float*)d_in[9];
  const float* Wo = (const float*)d_in[10];
  const float* bo = (const float*)d_in[11];

  const size_t perTensor = (size_t)BATCH * NUM_HEADS * SEQ * D_K;  // 4,194,304
  __bf16* Qbf = (__bf16*)d_ws;
  __bf16* Kbf = Qbf + perTensor;
  __bf16* Vtb = Kbf + perTensor;
  __bf16* Ctx = Vtb + perTensor;  // [B][S][D_MODEL] bf16

  dim3 gG(D_MODEL / 64, MROWS / 128);
  dim3 blk(256);

  proj_kernel<0, false><<<gG, blk, 0, stream>>>(query, nullptr, Wq, bq, Qbf, nullptr);
  proj_kernel<0, false><<<gG, blk, 0, stream>>>(key,   nullptr, Wk, bk, Kbf, nullptr);
  proj_kernel<1, false><<<gG, blk, 0, stream>>>(value, nullptr, Wv, bv, Vtb, nullptr);

  attn_kernel<<<dim3(SEQ / 128, BATCH * NUM_HEADS), blk, 0, stream>>>(
      Qbf, Kbf, Vtb, mask, Ctx);

  proj_kernel<2, true><<<gG, blk, 0, stream>>>(nullptr, Ctx, Wo, bo, nullptr,
                                               (float*)d_out);
}
